// KNRM_3049426780444
// MI455X (gfx1250) — compile-verified, matrix operand order checked by reading, not run on hardware
//
#include <hip/hip_runtime.h>

typedef __attribute__((ext_vector_type(16))) _Float16 v16h;
typedef __attribute__((ext_vector_type(8)))  _Float16 v8h;
typedef __attribute__((ext_vector_type(8)))  float    v8f;

#define BATCH 2048
#define LQ    32
#define LD    256
#define EMB   50
#define KPAD  64          // K padded to 2 x 32 for wmma_f32_16x16x32_f16
#define NK    21

#define EXP_M1 0.36787944117144233f   // e^-1 (ratio-of-ratios of adjacent RBF kernels)

__global__ __launch_bounds__(256)
void knrm_pair_kernel(const float* __restrict__ emb,
                      const float* __restrict__ w0, const float* __restrict__ b0,
                      const float* __restrict__ w1, const float* __restrict__ b1,
                      const float* __restrict__ w2, const float* __restrict__ b2,
                      const int* __restrict__ q1, const int* __restrict__ d1,
                      const int* __restrict__ q2, const int* __restrict__ d2,
                      float* __restrict__ out)
{
    __shared__ __align__(32) _Float16 sQ[LQ * KPAD];   // 4 KB
    __shared__ __align__(32) _Float16 sD[LD * KPAD];   // 32 KB
    __shared__ __align__(16) float    sMM[LQ * LD];    // 32 KB
    __shared__ float sKO[2 * NK];

    const int b    = blockIdx.x;
    const int tid  = threadIdx.x;
    const int lane = tid & 31;
    const int wave = tid >> 5;

    if (tid < 2 * NK) sKO[tid] = 0.0f;

    for (int p = 0; p < 2; ++p) {
        const int* qi = p ? q2 : q1;
        const int* di = p ? d2 : d1;

        __syncthreads();

        // ---- Stage 1: gather embeddings, L2-normalize, f32->f16 into LDS ----
        if (tid < LQ) {
            const int idx = qi[b * LQ + tid];
            const float* row = emb + (long)idx * EMB;
            float ss = 0.0f;
            #pragma unroll 10
            for (int k = 0; k < EMB; ++k) { float v = row[k]; ss += v * v; }
            const float inv = 1.0f / (sqrtf(ss) + 1e-13f);
            #pragma unroll
            for (int k = 0; k < KPAD; ++k)
                sQ[tid * KPAD + k] = (k < EMB) ? (_Float16)(row[k] * inv) : (_Float16)0.0f;
        }
        {
            const int idx = di[b * LD + tid];
            const float* row = emb + (long)idx * EMB;
            float ss = 0.0f;
            #pragma unroll 10
            for (int k = 0; k < EMB; ++k) { float v = row[k]; ss += v * v; }
            const float inv = 1.0f / (sqrtf(ss) + 1e-13f);
            #pragma unroll
            for (int k = 0; k < KPAD; ++k)
                sD[tid * KPAD + k] = (k < EMB) ? (_Float16)(row[k] * inv) : (_Float16)0.0f;
        }

        __syncthreads();

        // ---- Stage 2: mm[32,256] = qn * dn^T via WMMA (8 waves x 4 tiles) ----
        // A layout (16-bit 16x32): lane -> M = lane&15; per lane two 8-half
        // contiguous K chunks at kb=(lane>>4)*8 and kb+16.  B mirrors with N.
        {
            const int r16 = lane & 15;
            const int kb  = (lane >> 4) * 8;
            #pragma unroll
            for (int mt = 0; mt < 2; ++mt) {
                const int mrow = mt * 16 + r16;
                #pragma unroll
                for (int nt2 = 0; nt2 < 2; ++nt2) {
                    const int ntile = wave * 2 + nt2;
                    const int nrow  = ntile * 16 + r16;

                    v16h a, bmat;
                    // K = 0..31
                    *(v8h*)&a          = *(const v8h*)(sQ + mrow * KPAD + kb);
                    *((v8h*)&a + 1)    = *(const v8h*)(sQ + mrow * KPAD + kb + 16);
                    *(v8h*)&bmat       = *(const v8h*)(sD + nrow * KPAD + kb);
                    *((v8h*)&bmat + 1) = *(const v8h*)(sD + nrow * KPAD + kb + 16);
                    v8f c = {};
                    c = __builtin_amdgcn_wmma_f32_16x16x32_f16(
                            false, a, false, bmat, (short)0, c, false, false);
                    // K = 32..63 (rows 50..63 are zero-padded)
                    *(v8h*)&a          = *(const v8h*)(sQ + mrow * KPAD + 32 + kb);
                    *((v8h*)&a + 1)    = *(const v8h*)(sQ + mrow * KPAD + 32 + kb + 16);
                    *(v8h*)&bmat       = *(const v8h*)(sD + nrow * KPAD + 32 + kb);
                    *((v8h*)&bmat + 1) = *(const v8h*)(sD + nrow * KPAD + 32 + kb + 16);
                    c = __builtin_amdgcn_wmma_f32_16x16x32_f16(
                            false, a, false, bmat, (short)0, c, false, false);

                    // C/D layout: VGPR j holds M = j (lanes 0-15) / j+8 (16-31)
                    const int mbase = mt * 16 + (lane >> 4) * 8;
                    const int ncol  = ntile * 16 + r16;
                    #pragma unroll
                    for (int j = 0; j < 8; ++j)
                        sMM[(mbase + j) * LD + ncol] = c[j];
                }
            }
        }

        __syncthreads();

        // ---- Stage 3: Gaussian kernel pooling via exp-recurrence ----
        // 8 lanes own one row (32 cols each). mus are arithmetic (step 0.1),
        // so: g0 = exp(-50(x+0.95)^2), r0 = exp(10x+9),
        //     g_{k+1} = g_k * r_k,  r_{k+1} = r_k * e^-1.
        // 2 exps + 2 muls/kernel instead of 21 exps per element.
        {
            const int sub = lane >> 3;            // row group within wave (0..3)
            const int c0  = lane & 7;             // column phase within row
            const int q   = wave * 4 + sub;

            float acc[NK];
            #pragma unroll
            for (int m = 0; m < NK; ++m) acc[m] = 0.0f;

            #pragma unroll 4
            for (int i = 0; i < 32; ++i) {
                // rotate start column by 16*sub: 32 lanes -> 32 distinct banks
                const int col = (c0 + 8 * i + 16 * sub) & (LD - 1);
                const float x = sMM[q * LD + col];

                float g = __expf(-50.0f * (x + 0.95f) * (x + 0.95f));
                float r = __expf(10.0f * x + 9.0f);
                #pragma unroll
                for (int m = 0; m < NK - 1; ++m) {
                    acc[m] += g;
                    g *= r;
                    r *= EXP_M1;
                }
                const float d = x - 1.0f;          // exact-match kernel, sigma=1e-3
                acc[NK - 1] += __expf(-(d * d) * 500000.0f);
            }

            #pragma unroll
            for (int m = 0; m < NK; ++m) {
                float s = acc[m];
                s += __shfl_xor(s, 1, 32);
                s += __shfl_xor(s, 2, 32);
                s += __shfl_xor(s, 4, 32);
                if (c0 == 0)
                    atomicAdd(&sKO[p * NK + m], __logf(1.0f + s));
            }
        }
    }

    __syncthreads();

    // ---- Stage 4: tiny MLP (21->10->5->1) x2 + pairwise sigmoid ----
    if (tid == 0) {
        float logits[2];
        #pragma unroll
        for (int p = 0; p < 2; ++p) {
            float h0[10];
            #pragma unroll
            for (int o = 0; o < 10; ++o) {
                float s = b0[o];
                #pragma unroll
                for (int i = 0; i < NK; ++i) s += w0[o * NK + i] * sKO[p * NK + i];
                h0[o] = fmaxf(s, 0.0f);
            }
            float h1[5];
            #pragma unroll
            for (int o = 0; o < 5; ++o) {
                float s = b1[o];
                #pragma unroll
                for (int i = 0; i < 10; ++i) s += w1[o * 10 + i] * h0[i];
                h1[o] = fmaxf(s, 0.0f);
            }
            float s = b2[0];
            #pragma unroll
            for (int i = 0; i < 5; ++i) s += w2[i] * h1[i];
            logits[p] = s;
        }
        out[b] = 1.0f / (1.0f + __expf(-(logits[0] - logits[1])));
    }
}

extern "C" void kernel_launch(void* const* d_in, const int* in_sizes, int n_in,
                              void* d_out, int out_size, void* d_ws, size_t ws_size,
                              hipStream_t stream) {
    (void)in_sizes; (void)n_in; (void)d_ws; (void)ws_size; (void)out_size;
    // setup_inputs order: emb, w0, b0, w1, b1, w2, b2, query1, doc1, query2, doc2
    const float* emb = (const float*)d_in[0];
    const float* w0  = (const float*)d_in[1];
    const float* b0  = (const float*)d_in[2];
    const float* w1  = (const float*)d_in[3];
    const float* b1  = (const float*)d_in[4];
    const float* w2  = (const float*)d_in[5];
    const float* b2  = (const float*)d_in[6];
    const int*   q1  = (const int*)d_in[7];
    const int*   d1  = (const int*)d_in[8];
    const int*   q2  = (const int*)d_in[9];
    const int*   d2  = (const int*)d_in[10];
    float* out = (float*)d_out;

    knrm_pair_kernel<<<BATCH, 256, 0, stream>>>(emb, w0, b0, w1, b1, w2, b2,
                                                q1, d1, q2, d2, out);
}
